// Gemma4MoE_12326556139557
// MI455X (gfx1250) — compile-verified, matrix operand order checked by reading, not run on hardware
//
#include <hip/hip_runtime.h>
#include <hip/hip_bf16.h>

typedef __attribute__((ext_vector_type(16))) __bf16 v16bf;
typedef __attribute__((ext_vector_type(8)))  __bf16 v8bf;
typedef __attribute__((ext_vector_type(8)))  float  v8f;
typedef int v4i __attribute__((vector_size(16)));   // matches builtin's expected param type

#define T_TOK 1024
#define H_DIM 1024
#define E_EXP 16
#define I_DIM 1024
#define TOPK  2
#define BN    128   // 8 waves * 16 cols per block

// ---- CDNA5 async global->LDS path (guarded; falls back to sync copy) ----
#if defined(__has_builtin)
#if __has_builtin(__builtin_amdgcn_global_load_async_to_lds_b128)
#define HAVE_ASYNC_LDS 1
#endif
#endif
#ifndef HAVE_ASYNC_LDS
#define HAVE_ASYNC_LDS 0
#endif

__device__ __forceinline__ void stage16B(const float* gsrc, float* ldst) {
#if HAVE_ASYNC_LDS
    __builtin_amdgcn_global_load_async_to_lds_b128((v4i*)gsrc, (v4i*)ldst, 0, 0);
#else
    *(float4*)ldst = *(const float4*)gsrc;
#endif
}

__device__ __forceinline__ void wait_async0() {
#if HAVE_ASYNC_LDS
#if __has_builtin(__builtin_amdgcn_s_wait_asynccnt)
    __builtin_amdgcn_s_wait_asynccnt(0);
#else
    asm volatile("s_wait_asynccnt 0x0" ::: "memory");
#endif
#endif
}

// ---------------- workspace layout (bytes) ----------------
// 0      counts    16 int
// 64     fill      16 int
// 128    offsets   17 int            (pad to 256)
// 256    topk_id   T*2 int    (8192)
// 8448   topk_w    T*2 float  (8192)
// 16640  row_token 2048 int   (8192)
// 24832  row_w     2048 float (8192)
// 33024  act       2048*1024 bf16 (4 MB)

__global__ void moe_init(float* __restrict__ out, int* __restrict__ hdr) {
    int gid = blockIdx.x * blockDim.x + threadIdx.x;
    const int n = T_TOK * H_DIM;
    for (int i = gid; i < n; i += gridDim.x * blockDim.x) out[i] = 0.0f;
    if (gid < 64) hdr[gid] = 0;   // zero counts + fill (+pad)
}

__global__ void moe_route_topk(const float* __restrict__ logits,
                               const float* __restrict__ scale,
                               int* __restrict__ counts,
                               int* __restrict__ topk_id,
                               float* __restrict__ topk_w) {
    int t = blockIdx.x * blockDim.x + threadIdx.x;
    if (t >= T_TOK) return;
    float p[E_EXP];
    float mx = -3.4e38f;
    #pragma unroll
    for (int e = 0; e < E_EXP; ++e) { p[e] = logits[t * E_EXP + e]; mx = fmaxf(mx, p[e]); }
    float sum = 0.0f;
    #pragma unroll
    for (int e = 0; e < E_EXP; ++e) { p[e] = __expf(p[e] - mx); sum += p[e]; }
    const float inv = 1.0f / sum;
    int e0 = 0; float p0 = p[0];
    #pragma unroll
    for (int e = 1; e < E_EXP; ++e) if (p[e] > p0) { p0 = p[e]; e0 = e; }
    int e1 = (e0 == 0) ? 1 : 0; float p1 = p[e1];
    #pragma unroll
    for (int e = 0; e < E_EXP; ++e) if (e != e0 && p[e] > p1) { p1 = p[e]; e1 = e; }
    p0 *= inv; p1 *= inv;
    float renorm = p0 + p1;
    renorm = (renorm > 0.0f) ? renorm : 1.0f;
    topk_id[t * TOPK + 0] = e0;
    topk_id[t * TOPK + 1] = e1;
    topk_w[t * TOPK + 0] = (p0 / renorm) * scale[e0];
    topk_w[t * TOPK + 1] = (p1 / renorm) * scale[e1];
    atomicAdd(&counts[e0], 1);
    atomicAdd(&counts[e1], 1);
}

__global__ void moe_scan(const int* __restrict__ counts, int* __restrict__ offsets) {
    if (threadIdx.x == 0 && blockIdx.x == 0) {
        int acc = 0;
        for (int e = 0; e < E_EXP; ++e) { offsets[e] = acc; acc += counts[e]; }
        offsets[E_EXP] = acc;
    }
}

__global__ void moe_scatter(const int* __restrict__ topk_id,
                            const float* __restrict__ topk_w,
                            const int* __restrict__ offsets,
                            int* __restrict__ fill,
                            int* __restrict__ row_token,
                            float* __restrict__ row_w) {
    int t = blockIdx.x * blockDim.x + threadIdx.x;
    if (t >= T_TOK) return;
    #pragma unroll
    for (int k = 0; k < TOPK; ++k) {
        int e = topk_id[t * TOPK + k];
        int slot = atomicAdd(&fill[e], 1);
        int row = offsets[e] + slot;
        row_token[row] = t;
        row_w[row] = topk_w[t * TOPK + k];
    }
}

// pack 16 f32 (two contiguous 8-runs) into a bf16 fragment register set
__device__ __forceinline__ v16bf pack16(float4 a, float4 b, float4 c, float4 d) {
    v16bf v;
    v[0]=(__bf16)a.x;  v[1]=(__bf16)a.y;  v[2]=(__bf16)a.z;  v[3]=(__bf16)a.w;
    v[4]=(__bf16)b.x;  v[5]=(__bf16)b.y;  v[6]=(__bf16)b.z;  v[7]=(__bf16)b.w;
    v[8]=(__bf16)c.x;  v[9]=(__bf16)c.y;  v[10]=(__bf16)c.z; v[11]=(__bf16)c.w;
    v[12]=(__bf16)d.x; v[13]=(__bf16)d.y; v[14]=(__bf16)d.z; v[15]=(__bf16)d.w;
    return v;
}

// GEMM1: per expert, gathered tokens -> act = gelu(x w1^T) * (x w3^T)  (bf16 out)
__global__ void __launch_bounds__(256)
moe_gemm1(const float* __restrict__ x,
          const float* __restrict__ w1,
          const float* __restrict__ w3,
          const int* __restrict__ offsets,
          const int* __restrict__ row_token,
          __bf16* __restrict__ act) {
    __shared__ float lds_a[2][16][36];   // stride 36 keeps 16B runs aligned
    __shared__ int   s_tok[16];
    const int e    = blockIdx.x / (I_DIM / BN);
    const int nt   = blockIdx.x % (I_DIM / BN);
    const int tid  = threadIdx.x;
    const int wave = tid >> 5;
    const int lane = tid & 31;
    const int n    = lane & 15;          // N column within wave tile / A row index
    const int k0   = (lane & 16) ? 8 : 0;
    const int colBase = nt * BN + wave * 16;
    const int rowOff  = offsets[e];
    const int rowEnd  = offsets[e + 1];

    const float* w1base = w1 + ((size_t)e * I_DIM + colBase + n) * H_DIM;
    const float* w3base = w3 + ((size_t)e * I_DIM + colBase + n) * H_DIM;

    // staging role: threads 0..127 each move one 16B chunk of the 16x32 tile
    const int sr = tid >> 3;             // row 0..15 (for tid<128)
    const int sc = (tid & 7) * 4;        // col 0,4,...,28

    for (int rowBase = rowOff; rowBase < rowEnd; rowBase += 16) {
        if (tid < 16) {
            int slot = rowBase + tid;
            s_tok[tid] = (slot < rowEnd) ? row_token[slot] : -1;
        }
        __syncthreads();

        int stok = (tid < 128) ? s_tok[sr] : -1;
        const float* xrow = x + (size_t)(stok < 0 ? 0 : stok) * H_DIM + sc;
        const bool do_stage = (tid < 128) && (stok >= 0);
        if (tid < 128 && stok < 0) {     // zero-fill padded rows once, both buffers
            float4 z = {0.f, 0.f, 0.f, 0.f};
            *(float4*)&lds_a[0][sr][sc] = z;
            *(float4*)&lds_a[1][sr][sc] = z;
        }
        if (do_stage) stage16B(xrow, &lds_a[0][sr][sc]);   // prime buffer 0
        wait_async0();
        __syncthreads();

        v8f acc1 = {}; v8f acc3 = {};
        for (int kb = 0; kb < H_DIM; kb += 32) {
            const int cur = (kb >> 5) & 1;
            if (kb + 32 < H_DIM && do_stage)               // overlap next-tile copy
                stage16B(xrow + kb + 32, &lds_a[cur ^ 1][sr][sc]);

            const float4 a0 = *(const float4*)&lds_a[cur][n][k0];
            const float4 a1 = *(const float4*)&lds_a[cur][n][k0 + 4];
            const float4 a2 = *(const float4*)&lds_a[cur][n][16 + k0];
            const float4 a3 = *(const float4*)&lds_a[cur][n][16 + k0 + 4];
            const v16bf af = pack16(a0, a1, a2, a3);

            const float* p1 = w1base + kb;
            const float* p3 = w3base + kb;
            __builtin_prefetch(p1 + 32, 0, 1);
            __builtin_prefetch(p3 + 32, 0, 1);
            const v16bf bf1 = pack16(*(const float4*)(p1 + k0),      *(const float4*)(p1 + k0 + 4),
                                     *(const float4*)(p1 + 16 + k0), *(const float4*)(p1 + 16 + k0 + 4));
            const v16bf bf3 = pack16(*(const float4*)(p3 + k0),      *(const float4*)(p3 + k0 + 4),
                                     *(const float4*)(p3 + 16 + k0), *(const float4*)(p3 + 16 + k0 + 4));

            acc1 = __builtin_amdgcn_wmma_f32_16x16x32_bf16(false, af, false, bf1, (short)0, acc1, false, false);
            acc3 = __builtin_amdgcn_wmma_f32_16x16x32_bf16(false, af, false, bf3, (short)0, acc3, false, false);

            wait_async0();       // our next-tile async writes have landed
            __syncthreads();     // all waves done reading cur / writing cur^1
        }
        // epilogue: exact GELU gate, write bf16 activation rows
        #pragma unroll
        for (int r = 0; r < 8; ++r) {
            int M = r + ((lane & 16) ? 8 : 0);
            int row = rowBase + M;
            if (row < rowEnd) {
                float g = acc1[r];
                float gl = 0.5f * g * (1.0f + erff(g * 0.70710678118654752f));
                act[(size_t)row * I_DIM + colBase + n] = (__bf16)(gl * acc3[r]);
            }
        }
        __syncthreads();         // protect s_tok before next row tile
    }
}

// GEMM2: y = act * w2^T, gate-weighted atomic scatter into out[t,h]
__global__ void __launch_bounds__(256)
moe_gemm2(const __bf16* __restrict__ act,
          const float* __restrict__ w2,
          const int* __restrict__ offsets,
          const int* __restrict__ row_token,
          const float* __restrict__ row_w,
          float* __restrict__ out) {
    const int e    = blockIdx.x / (H_DIM / BN);
    const int ht   = blockIdx.x % (H_DIM / BN);
    const int tid  = threadIdx.x;
    const int wave = tid >> 5;
    const int lane = tid & 31;
    const int n    = lane & 15;
    const int k0   = (lane & 16) ? 8 : 0;
    const int colBase = ht * BN + wave * 16;
    const int rowOff  = offsets[e];
    const int rowEnd  = offsets[e + 1];
    const int total   = offsets[E_EXP];

    const float* w2base = w2 + ((size_t)e * H_DIM + colBase + n) * I_DIM;

    for (int rowBase = rowOff; rowBase < rowEnd; rowBase += 16) {
        v8f acc = {};
        int arow = rowBase + n;
        if (arow >= total) arow = total - 1;   // clamp: stray rows only feed unwritten D rows
        const __bf16* abase = act + (size_t)arow * I_DIM;

        for (int kb = 0; kb < I_DIM; kb += 32) {
            const v8bf a0 = *(const v8bf*)(abase + kb + k0);
            const v8bf a1 = *(const v8bf*)(abase + kb + 16 + k0);
            v16bf af;
            #pragma unroll
            for (int i = 0; i < 8; ++i) { af[i] = a0[i]; af[8 + i] = a1[i]; }

            const float* p = w2base + kb;
            __builtin_prefetch(p + 32, 0, 1);
            const v16bf bf = pack16(*(const float4*)(p + k0),      *(const float4*)(p + k0 + 4),
                                    *(const float4*)(p + 16 + k0), *(const float4*)(p + 16 + k0 + 4));
            acc = __builtin_amdgcn_wmma_f32_16x16x32_bf16(false, af, false, bf, (short)0, acc, false, false);
        }
        #pragma unroll
        for (int r = 0; r < 8; ++r) {
            int M = r + ((lane & 16) ? 8 : 0);
            int row = rowBase + M;
            if (row < rowEnd) {
                int tok = row_token[row];
                atomicAdd(&out[(size_t)tok * H_DIM + colBase + n], row_w[row] * acc[r]);
            }
        }
    }
}

extern "C" void kernel_launch(void* const* d_in, const int* in_sizes, int n_in,
                              void* d_out, int out_size, void* d_ws, size_t ws_size,
                              hipStream_t stream) {
    const float* x      = (const float*)d_in[0];
    const float* logits = (const float*)d_in[1];
    const float* scale  = (const float*)d_in[2];
    const float* w1     = (const float*)d_in[3];
    const float* w3     = (const float*)d_in[4];
    const float* w2     = (const float*)d_in[5];
    // d_in[6] = top_k scalar (fixed 2)
    float* out = (float*)d_out;

    char* ws = (char*)d_ws;
    int*    counts    = (int*)(ws + 0);
    int*    fill      = (int*)(ws + 64);
    int*    offsets   = (int*)(ws + 128);
    int*    topk_id   = (int*)(ws + 256);
    float*  topk_w    = (float*)(ws + 8448);
    int*    row_token = (int*)(ws + 16640);
    float*  row_w     = (float*)(ws + 24832);
    __bf16* act       = (__bf16*)(ws + 33024);

    moe_init<<<256, 256, 0, stream>>>(out, (int*)ws);
    moe_route_topk<<<T_TOK / 256, 256, 0, stream>>>(logits, scale, counts, topk_id, topk_w);
    moe_scan<<<1, 32, 0, stream>>>(counts, offsets);
    moe_scatter<<<T_TOK / 256, 256, 0, stream>>>(topk_id, topk_w, offsets, fill, row_token, row_w);
    moe_gemm1<<<E_EXP * (I_DIM / BN), 256, 0, stream>>>(x, w1, w3, offsets, row_token, act);
    moe_gemm2<<<E_EXP * (H_DIM / BN), 256, 0, stream>>>(act, w2, offsets, row_token, row_w, out);
}
